// MultiHeadedAttention_46763603919527
// MI455X (gfx1250) — compile-verified, matrix operand order checked by reading, not run on hardware
//
#include <hip/hip_runtime.h>

// MHA for MI455X (gfx1250), wave32. f32 in/out; one f32->f16 conversion pass, then
// all GEMMs run f16 WMMA (v_wmma_f32_16x16x32_f16, f32 accum). Data movement:
//   - X tiles:  TENSOR_LOAD_TO_LDS (TDM, TENSORcnt) with pad feature for LDS stride
//   - W/Q/K tiles: global_load_async_to_lds_b128 (ASYNCcnt)
//   - fragments: ds_load_b128 per ISA 7.12.2 layout

#define Bsz 8
#define Ssz 512
#define Hsz 1024
#define NHsz 16
#define Dsz 64

typedef __attribute__((ext_vector_type(16))) _Float16 v16h;
typedef __attribute__((ext_vector_type(8)))  _Float16 v8h;
typedef __attribute__((ext_vector_type(4)))  _Float16 v4h;
typedef __attribute__((ext_vector_type(8)))  float    v8f;
typedef __attribute__((ext_vector_type(4)))  float    v4f;
typedef __attribute__((ext_vector_type(4)))  unsigned int v4u;
typedef __attribute__((ext_vector_type(4)))  int      v4i;
typedef __attribute__((ext_vector_type(8)))  int      v8i;

#if __has_builtin(__builtin_amdgcn_tensor_load_to_lds)
#define HAVE_TDM 1
#else
#define HAVE_TDM 0
#endif

// Async 16B global->LDS copy (VGLOBAL opcode 98, ASYNCcnt).
__device__ __forceinline__ void async_copy_b128(unsigned int lds_byte_off,
                                                const void* gaddr) {
  asm volatile("global_load_async_to_lds_b128 %0, %1, off"
               :: "v"(lds_byte_off), "v"(gaddr) : "memory");
}
__device__ __forceinline__ void wait_async0() {
  asm volatile("s_wait_asynccnt 0x0" ::: "memory");
}
__device__ __forceinline__ unsigned int lds_off(const void* p) {
  return (unsigned int)(size_t)p;   // flat LDS pointer low 32 bits = LDS offset
}

#if HAVE_TDM
// TDM: load a rows x 32-half 2D f16 tile into LDS with 16-half row padding
// (LDS row stride 48 halves). Issued once per workgroup (wave 0).
__device__ __forceinline__ void tdm_issue_tile(const _Float16* gsrc,
                                               unsigned int lds_byte_off,
                                               unsigned int tensor_w,
                                               unsigned int tensor_h,
                                               unsigned int tile_h) {
  unsigned long long ga = (unsigned long long)(size_t)gsrc;
  v4u g0 = { 1u,                                   // count=1, user descriptor
             lds_byte_off,                         // lds_addr
             (unsigned int)ga,                     // global_addr[31:0]
             (unsigned int)((ga >> 32) & 0x01FFFFFFu) | (2u << 30) }; // addr hi + type=2
  v8i g1 = { (int)((1u << 16) | (1u << 20) | (3u << 22) | (7u << 25)),
             // data_size=2B, pad_enable, pad_interval=16dw, pad_amount=8dw
             (int)((tensor_w & 0xFFFFu) << 16),    // tensor_dim0 lo16 (bits 63:48)
             (int)((tensor_w >> 16) | ((tensor_h & 0xFFFFu) << 16)),
             (int)((tensor_h >> 16) | (32u << 16)),// tile_dim0 = 32 halves
             (int)tile_h,                          // tile_dim1 (tile_dim2 = 0)
             (int)tensor_w,                        // tensor_dim0_stride lo32
             0, 0 };
  v4i gz = { 0, 0, 0, 0 };
#if defined(__clang_major__) && (__clang_major__ >= 23)
  v8i gz8 = { 0, 0, 0, 0, 0, 0, 0, 0 };
  __builtin_amdgcn_tensor_load_to_lds(g0, g1, gz, gz, gz8, 0);
#else
  __builtin_amdgcn_tensor_load_to_lds(g0, g1, gz, gz, 0);
#endif
}
#endif

// 16x32 f16 A/B fragment from an LDS tile stored [row][k] row-major (ISA 7.12.2).
__device__ __forceinline__ v16h load_frag(const _Float16* lds, int rowBase, int kBase,
                                          int stride, int lane) {
  const _Float16* p = lds + (rowBase + (lane & 15)) * stride + kBase + ((lane >> 4) << 3);
  v8h lo = *(const v8h*)(p);
  v8h hi = *(const v8h*)(p + 16);
  return __builtin_shufflevector(lo, hi, 0,1,2,3,4,5,6,7,8,9,10,11,12,13,14,15);
}

__device__ __forceinline__ v8f wmma_f16(v16h a, v16h b, v8f c) {
  return __builtin_amdgcn_wmma_f32_16x16x32_f16(false, a, false, b, (short)0, c,
                                                false, false);
}

// ---------------------------------------------------------------------------
// f32 -> f16 conversion pass (bandwidth-bound; halves all downstream traffic).
// ---------------------------------------------------------------------------
__global__ __launch_bounds__(256) void cvt_f16(const float* __restrict__ src,
                                               _Float16* __restrict__ dst, int n) {
  int i = (blockIdx.x * 256 + threadIdx.x) * 4;
  if (i < n) {
    v4f v = *(const v4f*)(src + i);
    v4h h = { (_Float16)v.x, (_Float16)v.y, (_Float16)v.z, (_Float16)v.w };
    *(v4h*)(dst + i) = h;
  }
}

// ---------------------------------------------------------------------------
// Y = X @ W^T + bias.  X:[4096,1024] f16, W:[1024,1024] f16.
// Block tile 128(M) x 64(N), 8 waves, 2x2 WMMA tiles per wave, K staged 32 wide.
// X tile via TDM (wave 0), W tile via async b128 copies.
// MODE 0: f16 scatter to [B,NH,S,D].  MODE 1: f32 plain [M,N].
// ---------------------------------------------------------------------------
template <int MODE>
__global__ __launch_bounds__(256) void gemm_f16(const _Float16* __restrict__ X,
                                                const _Float16* __restrict__ W,
                                                const float* __restrict__ bias,
                                                _Float16* __restrict__ Yh,
                                                float* __restrict__ Yf) {
  constexpr int XS = 48;                        // 96B row stride (TDM pad) 16B aligned
  __shared__ __align__(16) _Float16 Xs[128 * XS];
  __shared__ __align__(16) _Float16 Ws[64 * XS];
  const int t = threadIdx.x;
  const int lane = t & 31, w = t >> 5;
  const int wm = w >> 1, wn = w & 1;            // 4 x 2 wave grid
  const int m0 = blockIdx.y * 128;
  const int n0 = blockIdx.x * 64;
  const unsigned int xs0 = lds_off(Xs), ws0 = lds_off(Ws);

  v8f acc[2][2];
#pragma unroll
  for (int i = 0; i < 2; ++i)
#pragma unroll
    for (int j = 0; j < 2; ++j) acc[i][j] = (v8f){0, 0, 0, 0, 0, 0, 0, 0};

  for (int k0 = 0; k0 < Hsz; k0 += 32) {
    __syncthreads();
#if HAVE_TDM
    if (w == 0)                                  // X tile 128x32 via TDM
      tdm_issue_tile(X + (size_t)m0 * Hsz + k0, xs0, Hsz, (unsigned)(Bsz * Ssz), 128);
#else
    {
      int r = t >> 2, c = (t & 3) * 8;
      async_copy_b128(xs0 + (unsigned)(r * XS + c) * 2,
                      X + (size_t)(m0 + r) * Hsz + k0 + c);
      async_copy_b128(xs0 + (unsigned)((r + 64) * XS + c) * 2,
                      X + (size_t)(m0 + r + 64) * Hsz + k0 + c);
    }
#endif
    {
      int r = t >> 2, c = (t & 3) * 8;           // W tile 64x32 async
      async_copy_b128(ws0 + (unsigned)(r * XS + c) * 2,
                      W + (size_t)(n0 + r) * Hsz + k0 + c);
    }
    if (k0 + 32 < Hsz) {                         // warm L2 for next X tile
      int r = t >> 1;
      __builtin_prefetch(X + (size_t)(m0 + r) * Hsz + k0 + 32, 0, 1);
    }
    wait_async0();
#if HAVE_TDM
    if (w == 0) __builtin_amdgcn_s_wait_tensorcnt(0);
#endif
    __syncthreads();

    v16h a0 = load_frag(Xs, wm * 32, 0, XS, lane);
    v16h a1 = load_frag(Xs, wm * 32 + 16, 0, XS, lane);
    v16h b0 = load_frag(Ws, wn * 32, 0, XS, lane);
    v16h b1 = load_frag(Ws, wn * 32 + 16, 0, XS, lane);
    acc[0][0] = wmma_f16(a0, b0, acc[0][0]);
    acc[0][1] = wmma_f16(a0, b1, acc[0][1]);
    acc[1][0] = wmma_f16(a1, b0, acc[1][0]);
    acc[1][1] = wmma_f16(a1, b1, acc[1][1]);
  }

#pragma unroll
  for (int ti = 0; ti < 2; ++ti)
#pragma unroll
    for (int tj = 0; tj < 2; ++tj) {
      int mrow = m0 + wm * 32 + ti * 16 + ((lane >> 4) << 3);
      int ncol = n0 + wn * 32 + tj * 16 + (lane & 15);
      float bv = bias[ncol];
#pragma unroll
      for (int r = 0; r < 8; ++r) {
        float val = acc[ti][tj][r] + bv;
        int m = mrow + r;
        if (MODE == 0) {
          int bb = m >> 9, si = m & 511;        // m = b*512 + s
          int hh = ncol >> 6, dd = ncol & 63;   // n = h*64 + d
          Yh[(((size_t)(bb * NHsz + hh) * Ssz + si) * Dsz) + dd] = (_Float16)val;
        } else {
          Yf[(size_t)m * Hsz + ncol] = val;
        }
      }
    }
}

// ---------------------------------------------------------------------------
// Attention per (b, h, 32 query rows):
//   scores = QK^T/8 (WMMA, f16 LDS) -> f32 softmax(+mask)
//   probs' = probs*large(lena,c) + hy0*kg[b,0] + hy1*kg[b,1] -> f16
//   ctx    = probs' @ V (WMMA), written f16 [B,S,H]
// ---------------------------------------------------------------------------
__global__ __launch_bounds__(256) void attn_wmma(const _Float16* __restrict__ qb,
                                                 const _Float16* __restrict__ kb,
                                                 const _Float16* __restrict__ vb,
                                                 const float* __restrict__ mask,
                                                 const int* __restrict__ lena,
                                                 const float* __restrict__ kg,
                                                 const float* __restrict__ hy,
                                                 const int* __restrict__ layerp,
                                                 _Float16* __restrict__ ctx) {
  constexpr int QT = 32;
  constexpr int KS = 72;                         // 144B row stride: 16B aligned
  constexpr int PSTR = 520;                      // 1040B rows: 16B aligned
  __shared__ __align__(16) _Float16 Qs[QT * KS];
  __shared__ __align__(16) _Float16 KVs[64 * KS];
  __shared__ __align__(16) _Float16 Ps[QT * PSTR];
  __shared__ float red[QT][8];

  const int t = threadIdx.x;
  const int lane = t & 31, w = t >> 5;
  const int qt = blockIdx.x, h = blockIdx.y, b = blockIdx.z;
  const int q0 = qt * QT;
  const size_t head = (size_t)(b * NHsz + h) * Ssz * Dsz;
  const _Float16* qh = qb + head;
  const _Float16* kh = kb + head;
  const _Float16* vh = vb + head;
  const unsigned int qs0 = lds_off(Qs), kvs0 = lds_off(KVs);

  const int layer = layerp[0];
  const float hy0 = hy[layer * 3 + 0];
  const float hy1 = hy[layer * 3 + 1];
  const float cc  = hy[layer * 3 + 2];
  const int len = lena[b];

  // Q tile 32x64 halves = 256 x 16B chunks (async).
  {
    int r = t >> 3, c = (t & 7) * 8;
    async_copy_b128(qs0 + (unsigned)(r * KS + c) * 2,
                    qh + (size_t)(q0 + r) * Dsz + c);
  }

  const int mtile = (w & 1) * 16;                // 2 m-tiles
  const int ntile = (w >> 1) * 16;               // 4 n-tiles

  // ---- Phase 1: scores ----
  for (int j0 = 0; j0 < Ssz; j0 += 64) {
    __syncthreads();
    {                                            // K tile 64x64 halves, 2 chunks/thread
      int r = t >> 3, c = (t & 7) * 8;
      async_copy_b128(kvs0 + (unsigned)(r * KS + c) * 2,
                      kh + (size_t)(j0 + r) * Dsz + c);
      async_copy_b128(kvs0 + (unsigned)((r + 32) * KS + c) * 2,
                      kh + (size_t)(j0 + r + 32) * Dsz + c);
    }
    wait_async0();
    __syncthreads();
    v8f acc = (v8f){0, 0, 0, 0, 0, 0, 0, 0};
#pragma unroll
    for (int ks = 0; ks < 2; ++ks) {
      v16h a  = load_frag(Qs,  mtile, ks * 32, KS, lane);
      v16h bf = load_frag(KVs, ntile, ks * 32, KS, lane);
      acc = wmma_f16(a, bf, acc);
    }
    int row = mtile + ((lane >> 4) << 3);
    int col = j0 + ntile + (lane & 15);
#pragma unroll
    for (int r = 0; r < 8; ++r)
      Ps[(row + r) * PSTR + col] = (_Float16)(acc[r] * 0.125f);
  }
  __syncthreads();

  // ---- Softmax + post-softmax modifiers (8 threads per row) ----
  {
    const int r = t >> 3;
    const int c8 = t & 7;
    const int rq = q0 + r;
    const float* mrow = mask + ((size_t)b * Ssz + rq) * Ssz;
    float lmax = -1e30f;
    for (int j = c8; j < Ssz; j += 8)
      lmax = fmaxf(lmax, (float)Ps[r * PSTR + j] + mrow[j]);
    red[r][c8] = lmax;
    __syncthreads();
    float rmax = red[r][0];
#pragma unroll
    for (int i = 1; i < 8; ++i) rmax = fmaxf(rmax, red[r][i]);
    __syncthreads();
    float lsum = 0.f;
    for (int j = c8; j < Ssz; j += 8)
      lsum += __expf((float)Ps[r * PSTR + j] + mrow[j] - rmax);
    red[r][c8] = lsum;
    __syncthreads();
    float rsum = 0.f;
#pragma unroll
    for (int i = 0; i < 8; ++i) rsum += red[r][i];
    const float inv = 1.0f / rsum;
    const bool rowa_r = (rq >= 1) && (rq < len - 1);
    const bool cola_r = (rq >= len) && (rq < Ssz - 1);
    const float* kg0 = kg + ((size_t)(b * 2 + 0) * Ssz + rq) * Ssz;
    const float* kg1 = kg + ((size_t)(b * 2 + 1) * Ssz + rq) * Ssz;
    for (int j = c8; j < Ssz; j += 8) {
      float s = (float)Ps[r * PSTR + j] + mrow[j];
      float p = __expf(s - rmax) * inv;
      bool rowa_j = (j >= 1) && (j < len - 1);
      bool cola_j = (j >= len) && (j < Ssz - 1);
      bool onblk = (rowa_r && cola_j) || (rowa_j && cola_r);
      float lg = onblk ? cc : 1.0f;
      Ps[r * PSTR + j] = (_Float16)(p * lg + hy0 * kg0[j] + hy1 * kg1[j]);
    }
  }
  __syncthreads();

  // ---- Phase 2: ctx = P @ V (V staged transposed [d][j]) ----
  v8f oacc = (v8f){0, 0, 0, 0, 0, 0, 0, 0};
  for (int j0 = 0; j0 < Ssz; j0 += 64) {
    __syncthreads();
    for (int i = t; i < 64 * Dsz; i += 256) {
      int r = i >> 6, c = i & 63;
      KVs[c * KS + r] = vh[(size_t)(j0 + r) * Dsz + c];
    }
    __syncthreads();
#pragma unroll
    for (int ks = 0; ks < 2; ++ks) {
      v16h a  = load_frag(Ps,  mtile, j0 + ks * 32, PSTR, lane);
      v16h bf = load_frag(KVs, ntile, ks * 32, KS, lane);
      oacc = wmma_f16(a, bf, oacc);
    }
  }
  {
    int row = q0 + mtile + ((lane >> 4) << 3);
    int dcol = ntile + (lane & 15);
#pragma unroll
    for (int r = 0; r < 8; ++r)
      ctx[((size_t)b * Ssz + row + r) * Hsz + h * Dsz + dcol] = oacc[r];
  }
}

extern "C" void kernel_launch(void* const* d_in, const int* in_sizes, int n_in,
                              void* d_out, int out_size, void* d_ws, size_t ws_size,
                              hipStream_t stream) {
  (void)in_sizes; (void)n_in; (void)out_size; (void)ws_size;
  const float* query = (const float*)d_in[0];
  const float* keyi  = (const float*)d_in[1];
  const float* value = (const float*)d_in[2];
  const float* mask  = (const float*)d_in[3];
  const int*   lena  = (const int*)d_in[4];
  const float* kg    = (const float*)d_in[5];
  const float* hy    = (const float*)d_in[6];
  const int*   layer = (const int*)d_in[7];
  const float* Wq = (const float*)d_in[8];
  const float* bq = (const float*)d_in[9];
  const float* Wk = (const float*)d_in[10];
  const float* bk = (const float*)d_in[11];
  const float* Wv = (const float*)d_in[12];
  const float* bv = (const float*)d_in[13];
  const float* Wo = (const float*)d_in[14];
  const float* bo = (const float*)d_in[15];
  float* out = (float*)d_out;

  const size_t ACT = (size_t)Bsz * Ssz * Hsz;   // 4 Mi
  const size_t WSZ = (size_t)Hsz * Hsz;         // 1 Mi
  _Float16* hws = (_Float16*)d_ws;
  _Float16* xq = hws;                            // f16 copies of inputs
  _Float16* xk = xq + ACT;
  _Float16* xv = xk + ACT;
  _Float16* wq = xv + ACT;                       // f16 weights
  _Float16* wk = wq + WSZ;
  _Float16* wv = wk + WSZ;
  _Float16* wo = wv + WSZ;
  _Float16* qp = wo + WSZ;                       // projected heads [B,NH,S,D]
  _Float16* kp = qp + ACT;
  _Float16* vp = kp + ACT;
  _Float16* cx = vp + ACT;                       // context [B,S,H]

  const int cb_act = (int)(ACT / 4 / 256);       // 4096 blocks
  const int cb_w   = (int)(WSZ / 4 / 256);       // 1024 blocks
  cvt_f16<<<cb_act, 256, 0, stream>>>(query, xq, (int)ACT);
  cvt_f16<<<cb_act, 256, 0, stream>>>(keyi,  xk, (int)ACT);
  cvt_f16<<<cb_act, 256, 0, stream>>>(value, xv, (int)ACT);
  cvt_f16<<<cb_w,   256, 0, stream>>>(Wq, wq, (int)WSZ);
  cvt_f16<<<cb_w,   256, 0, stream>>>(Wk, wk, (int)WSZ);
  cvt_f16<<<cb_w,   256, 0, stream>>>(Wv, wv, (int)WSZ);
  cvt_f16<<<cb_w,   256, 0, stream>>>(Wo, wo, (int)WSZ);

  dim3 pgrid(Hsz / 64, (Bsz * Ssz) / 128);       // (16, 32)
  dim3 agrid(Ssz / 32, NHsz, Bsz);               // (16, 16, 8)
  gemm_f16<0><<<pgrid, 256, 0, stream>>>(xq, wq, bq, qp, nullptr);
  gemm_f16<0><<<pgrid, 256, 0, stream>>>(xk, wk, bk, kp, nullptr);
  gemm_f16<0><<<pgrid, 256, 0, stream>>>(xv, wv, bv, vp, nullptr);
  attn_wmma<<<agrid, 256, 0, stream>>>(qp, kp, vp, mask, lena, kg, hy, layer, cx);
  gemm_f16<1><<<pgrid, 256, 0, stream>>>(cx, wo, bo, nullptr, out);
}
